// AddNormAndReduce_49091476194126
// MI455X (gfx1250) — matrix operand order repaired and checked by reading, not compiled, not run on hardware
//
#include <hip/hip_runtime.h>

// ---------------------------------------------------------------------------
// Fused residual-add + LayerNorm(W) + 1x1 conv (channel GEMM) + ReLU
// for MI455X (gfx1250), bf16x3 split-precision WMMA, fp32 accumulate.
//
// Shapes (hardcoded, match reference setup): B=4, C=256, H=256, W=256, O=256.
// Grid: one workgroup per (b,h) pair -> 1024 WGs of 512 threads (16 waves).
// Wave i owns output o-tile i (16 rows of O) x all 256 w (16 N-tiles).
// K (=C) processed in chunks of 32 (one v_wmma_f32_16x16x32_bf16 depth).
//
// Streaming traffic (x, y, out: 805 MB touched once) uses non-temporal
// loads/stores; next K-chunk rows are prefetched under the WMMA phase.
// Tile loop processes pairs of N-tiles with interleaved WMMAs so dependent
// same-accumulator ops alternate with independent ones and ds_loads stay
// two tiles ahead of their consumers.
// ---------------------------------------------------------------------------

typedef __attribute__((ext_vector_type(16))) __bf16 bf16x16;
typedef __attribute__((ext_vector_type(8)))  float  f32x8;
typedef __attribute__((ext_vector_type(4)))  float  f32x4;

#define B_DIM 4
#define C_DIM 256
#define H_DIM 256
#define W_DIM 256
#define O_DIM 256
#define KCH   32
#define LDS_STRIDE 40   // bf16 units per w-row (32 used + 8 pad) -> 80B, 16B aligned

#define WMMA_BF16(A, Bm, Cm) \
    __builtin_amdgcn_wmma_f32_16x16x32_bf16(false, (A), false, (Bm), (short)0, (Cm), false, false)

__device__ __forceinline__ bf16x16 load_frag_global(const __bf16* p) {
    // p points at this lane's 16 consecutive-K bf16 values (16B aligned);
    // second half of the fragment is 8 elements (16B) later (K + 8).
    bf16x16 f;
    reinterpret_cast<uint4*>(&f)[0] = reinterpret_cast<const uint4*>(p)[0];
    reinterpret_cast<uint4*>(&f)[1] = *reinterpret_cast<const uint4*>(p + 8);
    return f;
}

__device__ __forceinline__ bf16x16 load_frag_lds(const __bf16* p) {
    // 16 consecutive bf16 (32B) from LDS, 16B aligned -> 2x ds_load_b128.
    bf16x16 f;
    reinterpret_cast<uint4*>(&f)[0] = reinterpret_cast<const uint4*>(p)[0];
    reinterpret_cast<uint4*>(&f)[1] = reinterpret_cast<const uint4*>(p)[1];
    return f;
}

// Prepack conv weights [O x C] fp32 into bf16 hi + bf16 lo (row-major, C fast).
__global__ void prep_weights_kernel(const float* __restrict__ wsrc,
                                    __bf16* __restrict__ whi,
                                    __bf16* __restrict__ wlo) {
    int i = blockIdx.x * blockDim.x + threadIdx.x;  // 0 .. 65535
    float v = wsrc[i];
    __bf16 h = (__bf16)v;            // RNE
    whi[i] = h;
    wlo[i] = (__bf16)(v - (float)h); // residual
}

__global__ __launch_bounds__(512, 1)
void fused_addln_conv_relu_kernel(const float* __restrict__ x,
                                  const float* __restrict__ y,
                                  const float* __restrict__ lng,
                                  const float* __restrict__ lnb,
                                  const __bf16* __restrict__ whi,
                                  const __bf16* __restrict__ wlo,
                                  float* __restrict__ out) {
    __shared__ __bf16 sHi[W_DIM * LDS_STRIDE];   // normed hi, layout [w][c_local]
    __shared__ __bf16 sLo[W_DIM * LDS_STRIDE];   // normed lo

    const int tid  = threadIdx.x;
    const int lane = tid & 31;
    const int wave = tid >> 5;                   // 0..15
    const int b    = blockIdx.x >> 8;            // (b,h) per workgroup
    const int h    = blockIdx.x & 255;

    // Each lane covers w positions {4*lane+k, 128+4*lane+k : k=0..3} in the LN
    // phase; cache the LN affine params for those positions once.
    const f32x4 gq0 = *reinterpret_cast<const f32x4*>(lng + lane * 4);
    const f32x4 gq1 = *reinterpret_cast<const f32x4*>(lng + 128 + lane * 4);
    const f32x4 bq0 = *reinterpret_cast<const f32x4*>(lnb + lane * 4);
    const f32x4 bq1 = *reinterpret_cast<const f32x4*>(lnb + 128 + lane * 4);

    f32x8 acc[16];
    #pragma unroll
    for (int t = 0; t < 16; ++t) { f32x8 z = {}; acc[t] = z; }

    const int obase = wave * 16;                 // this wave's 16 output rows
    const int arow  = obase + (lane & 15);       // A-fragment: lane = M row
    const int asel  = (lane >> 4) & 1;           // lane half selects K sub-runs
    const int bn    = lane & 15;                 // B-fragment: lane = N column
    const int bkh   = lane >> 4;                 // lane half selects K half

    for (int kc = 0; kc < C_DIM; kc += KCH) {
        // ---- phase 1: residual add + LayerNorm for 32 channel rows --------
        __syncthreads();   // previous chunk's LDS reads complete
        #pragma unroll
        for (int r = 0; r < 2; ++r) {
            const int cl = wave * 2 + r;         // local K index 0..31
            const int c  = kc + cl;
            const size_t rowoff = (((size_t)b * C_DIM + c) * H_DIM + h) * (size_t)W_DIM;
            const f32x4 xa = __builtin_nontemporal_load(
                reinterpret_cast<const f32x4*>(x + rowoff + lane * 4));
            const f32x4 xb = __builtin_nontemporal_load(
                reinterpret_cast<const f32x4*>(x + rowoff + 128 + lane * 4));
            const f32x4 ya = __builtin_nontemporal_load(
                reinterpret_cast<const f32x4*>(y + rowoff + lane * 4));
            const f32x4 yb = __builtin_nontemporal_load(
                reinterpret_cast<const f32x4*>(y + rowoff + 128 + lane * 4));
            float v[8];
            v[0] = xa.x + ya.x; v[1] = xa.y + ya.y; v[2] = xa.z + ya.z; v[3] = xa.w + ya.w;
            v[4] = xb.x + yb.x; v[5] = xb.y + yb.y; v[6] = xb.z + yb.z; v[7] = xb.w + yb.w;
            float s = 0.f, s2 = 0.f;
            #pragma unroll
            for (int j = 0; j < 8; ++j) { s += v[j]; s2 += v[j] * v[j]; }
            #pragma unroll
            for (int off = 16; off > 0; off >>= 1) {
                s  += __shfl_xor(s,  off, 32);
                s2 += __shfl_xor(s2, off, 32);
            }
            const float mean = s * (1.0f / 256.0f);
            const float var  = s2 * (1.0f / 256.0f) - mean * mean;
            const float rstd = rsqrtf(var + 1e-5f);
            const float gg[8] = {gq0.x, gq0.y, gq0.z, gq0.w, gq1.x, gq1.y, gq1.z, gq1.w};
            const float bb[8] = {bq0.x, bq0.y, bq0.z, bq0.w, bq1.x, bq1.y, bq1.z, bq1.w};
            #pragma unroll
            for (int j = 0; j < 8; ++j) {
                const int w = (j >> 2) * 128 + lane * 4 + (j & 3);
                const float nv = (v[j] - mean) * rstd * gg[j] + bb[j];
                const __bf16 hi = (__bf16)nv;
                const __bf16 lo = (__bf16)(nv - (float)hi);
                sHi[w * LDS_STRIDE + cl] = hi;
                sLo[w * LDS_STRIDE + cl] = lo;
            }
        }
        __syncthreads();   // LDS tile visible to all waves

        // ---- phase 2: WMMA over this K chunk ------------------------------
        // A fragment (16x32 bf16): lanes 0-15 hold K {0..15},
        // lanes 16-31 hold K {16..31} of rows obase+lane%16.
        const __bf16* pah = whi + (size_t)arow * C_DIM + kc + asel * 16;
        const __bf16* pal = wlo + (size_t)arow * C_DIM + kc + asel * 16;
        const bf16x16 ahi = load_frag_global(pah);
        const bf16x16 alo = load_frag_global(pal);

        // Prefetch next chunk's x/y rows (hidden under the WMMAs below) so the
        // next memory phase hits cache instead of exposing HBM latency.
        if (kc + KCH < C_DIM) {
            #pragma unroll
            for (int r = 0; r < 2; ++r) {
                const int cn = kc + KCH + wave * 2 + r;
                const size_t ro = (((size_t)b * C_DIM + cn) * H_DIM + h) * (size_t)W_DIM;
                __builtin_prefetch(x + ro + lane * 4, 0, 1);
                __builtin_prefetch(x + ro + 128 + lane * 4, 0, 1);
                __builtin_prefetch(y + ro + lane * 4, 0, 1);
                __builtin_prefetch(y + ro + 128 + lane * 4, 0, 1);
            }
        }

        // Two N-tiles per iteration: 8 ds_load_b128 in flight, WMMAs
        // interleaved across the two independent accumulators.
        #pragma unroll
        for (int t = 0; t < 16; t += 2) {
            const int row0 = ((t + 0) * 16 + bn) * LDS_STRIDE + bkh * 16;
            const int row1 = ((t + 1) * 16 + bn) * LDS_STRIDE + bkh * 16;
            const bf16x16 bhi0 = load_frag_lds(&sHi[row0]);
            const bf16x16 blo0 = load_frag_lds(&sLo[row0]);
            const bf16x16 bhi1 = load_frag_lds(&sHi[row1]);
            const bf16x16 blo1 = load_frag_lds(&sLo[row1]);
            acc[t + 0] = WMMA_BF16(ahi, bhi0, acc[t + 0]);
            acc[t + 1] = WMMA_BF16(ahi, bhi1, acc[t + 1]);
            acc[t + 0] = WMMA_BF16(alo, bhi0, acc[t + 0]);
            acc[t + 1] = WMMA_BF16(alo, bhi1, acc[t + 1]);
            acc[t + 0] = WMMA_BF16(ahi, blo0, acc[t + 0]);
            acc[t + 1] = WMMA_BF16(ahi, blo1, acc[t + 1]);
        }
    }

    // ---- epilogue: ReLU + store (streaming, non-temporal). C/D layout:
    // elem v in lane l is M = v + 8*(l>>4), N = l&15; o = obase+M, w = t*16+N.
    const int orow0 = obase + 8 * (lane >> 4);
    #pragma unroll
    for (int t = 0; t < 16; ++t) {
        const int w = t * 16 + (lane & 15);
        float* po = out + (((size_t)b * O_DIM + orow0) * H_DIM + h) * (size_t)W_DIM + w;
        #pragma unroll
        for (int v = 0; v < 8; ++v) {
            const float val = acc[t][v];
            __builtin_nontemporal_store(val > 0.0f ? val : 0.0f,
                                        po + (size_t)v * (H_DIM * W_DIM));
        }
    }
}

extern "C" void kernel_launch(void* const* d_in, const int* in_sizes, int n_in,
                              void* d_out, int out_size, void* d_ws, size_t ws_size,
                              hipStream_t stream) {
    (void)in_sizes; (void)n_in; (void)out_size; (void)ws_size;
    const float* x   = (const float*)d_in[0];
    const float* y   = (const float*)d_in[1];
    const float* lng = (const float*)d_in[2];
    const float* lnb = (const float*)d_in[3];
    const float* cw  = (const float*)d_in[4];

    __bf16* whi = (__bf16*)d_ws;                 // 65536 bf16 = 128 KB
    __bf16* wlo = whi + O_DIM * C_DIM;           // +128 KB (ws use: 256 KB)

    prep_weights_kernel<<<dim3(O_DIM * C_DIM / 256), dim3(256), 0, stream>>>(cw, whi, wlo);

    fused_addln_conv_relu_kernel<<<dim3(B_DIM * H_DIM), dim3(512), 0, stream>>>(
        x, y, lng, lnb, whi, wlo, (float*)d_out);
}
